// TransFusionHead_13391708028997
// MI455X (gfx1250) — compile-verified, matrix operand order checked by reading, not run on hardware
//
#include <hip/hip_runtime.h>
#include <hip/hip_bf16.h>
#include <cstdint>
#include <cstddef>

// ---------------------------------------------------------------------------
// Types / constants
// ---------------------------------------------------------------------------
typedef __attribute__((ext_vector_type(16))) _Float16     v16h;
typedef __attribute__((ext_vector_type(8)))  float        v8f;
typedef __attribute__((ext_vector_type(4)))  unsigned int u32x4;
typedef __attribute__((ext_vector_type(8)))  int          i32x8;
typedef __attribute__((ext_vector_type(4)))  int          i32x4;

#define NB      2
#define CINC    512
#define IH      180
#define IW      180
#define HW      32400      // IH*IW
#define HIDC    128
#define NCLS    10
#define NQ      200        // P
#define NHEAD   8
#define FFND    256
#define HHID    64
#define NQP     208        // queries padded to 16
#define QTILES  13         // NQP/16
#define SA_NKP  224        // self-attn keys padded to 32
#define CA_NKP  32416      // cross-attn keys padded to 32
#define CA_SPLITS 16
#define CONV_NT 3          // n-tiles per wave in conv (48 pixels / block)

#if __has_builtin(__builtin_amdgcn_tensor_load_to_lds) && \
    __has_builtin(__builtin_amdgcn_s_wait_tensorcnt)
#define HAVE_TDM 1
#else
#define HAVE_TDM 0
#endif

// ---------------------------------------------------------------------------
// WMMA helpers.  ISA 7.12.2 fragment layouts:
//  A (16x32 f16): lane L in [0,16): row = L; two contiguous 8-half runs at
//                 K = 8*half and 16+8*half (half = lane>>4).
//  B (32x16 f16): lane = output column, elems walk K with same run pattern.
//  C/D (16x16 f32): VGPR r holds row r (lanes 0-15) / row 8+r (lanes 16-31),
//                   col = lane & 15.
// ---------------------------------------------------------------------------
static __device__ __forceinline__ v16h load_frag(const _Float16* row32, int half) {
  v16h r;
  ((u32x4*)&r)[0] = *(const u32x4*)(row32 + 8 * half);
  ((u32x4*)&r)[1] = *(const u32x4*)(row32 + 16 + 8 * half);
  return r;
}
static __device__ __forceinline__ v8f wmma16(v16h a, v16h b, v8f c) {
  return __builtin_amdgcn_wmma_f32_16x16x32_f16(false, a, false, b, (short)0, c,
                                                false, false);
}
static __device__ __forceinline__ float rmax16(float v) {
#pragma unroll
  for (int o = 1; o < 16; o <<= 1) v = fmaxf(v, __shfl_xor(v, o, 16));
  return v;
}
static __device__ __forceinline__ float rsum16(float v) {
#pragma unroll
  for (int o = 1; o < 16; o <<= 1) v += __shfl_xor(v, o, 16);
  return v;
}
static __device__ __forceinline__ unsigned fkey(float f) {
  unsigned u = __float_as_uint(f);
  return (u & 0x80000000u) ? ~u : (u | 0x80000000u);
}

// ---------------------------------------------------------------------------
// f32 -> f16 convert
// ---------------------------------------------------------------------------
__global__ void k_cvt(const float* __restrict__ s, _Float16* __restrict__ d, int n) {
  int i = blockIdx.x * 256 + threadIdx.x;
  if (i < n) d[i] = (_Float16)s[i];
}

// ---------------------------------------------------------------------------
// Implicit-GEMM 3x3 SAME conv via WMMA.
// A[M, K=Cin*9] f16 (weights, row-major), B = im2col(X) staged in LDS.
// Block covers 48 output pixels (3 n-tiles per wave); wave w owns M-tile w.
// mode 0: +bias -> f32 + f16 out   (shared conv)
// mode 1: BN*s+b, ReLU -> f16 out  (hm conv1)
// mode 2: +bias, sigmoid -> f32 out, only m<NCLS (hm conv2)
// ---------------------------------------------------------------------------
__global__ void k_conv_wmma(const _Float16* __restrict__ X,  // [NB][Cin][IH][IW]
                            const _Float16* __restrict__ A,  // [M][Cin*9]
                            int Cin, int Mtiles, int mode,
                            const float* __restrict__ p0,
                            const float* __restrict__ p1,
                            float* __restrict__ oF32, _Float16* __restrict__ oF16) {
  const int K  = Cin * 9;
  const int b  = blockIdx.y;
  const int n0 = blockIdx.x * (16 * CONV_NT);
  const int tid = threadIdx.x;
  const int w = tid >> 5, lane = tid & 31, half = lane >> 4, ln = lane & 15;
  const _Float16* Xb = X + (size_t)b * Cin * HW;
  __shared__ __align__(16) _Float16 Bs[16 * CONV_NT * 32];  // [n][k] tile
  v8f acc[CONV_NT];
#pragma unroll
  for (int t = 0; t < CONV_NT; ++t) acc[t] = (v8f){};
  const int m0 = w * 16;
  for (int k0 = 0; k0 < K; k0 += 32) {
    __syncthreads();
    for (int e = tid; e < 16 * CONV_NT * 32; e += blockDim.x) {
      int n = e >> 5, kl = e & 31;
      int k = k0 + kl;
      int c = k / 9, rs = k - c * 9;
      int dy = rs / 3 - 1, dx = rs - (rs / 3) * 3 - 1;
      int p = n0 + n;
      int y = p / IW + dy, x = p - (p / IW) * IW + dx;
      _Float16 v = (_Float16)0.f;
      if (y >= 0 && y < IH && x >= 0 && x < IW) v = Xb[(size_t)c * HW + y * IW + x];
      Bs[n * 32 + kl] = v;
    }
    __syncthreads();
    if (w < Mtiles) {
      const _Float16* ar = A + (size_t)(m0 + ln) * K + k0;
      if (k0 + 32 < K) __builtin_prefetch(ar + 32, 0, 1);
      v16h af = load_frag(ar, half);
#pragma unroll
      for (int t = 0; t < CONV_NT; ++t) {
        v16h bf = load_frag(&Bs[(16 * t + ln) * 32], half);
        acc[t] = wmma16(af, bf, acc[t]);
      }
    }
  }
  if (w < Mtiles) {
#pragma unroll
    for (int t = 0; t < CONV_NT; ++t) {
#pragma unroll
      for (int r = 0; r < 8; ++r) {
        int m = m0 + r + 8 * half;
        int p = n0 + 16 * t + ln;
        float v = acc[t][r];
        if (mode == 0) {
          v += p0[m];
          size_t o = ((size_t)b * HIDC + m) * HW + p;
          oF32[o] = v;
          oF16[o] = (_Float16)v;
        } else if (mode == 1) {
          v = fmaxf(v * p0[m] + p1[m], 0.f);
          oF16[((size_t)b * HIDC + m) * HW + p] = (_Float16)v;
        } else {
          if (m < NCLS) {
            v += p0[m];
            oF32[((size_t)b * NCLS + m) * HW + p] = 1.f / (1.f + __expf(-v));
          }
        }
      }
    }
  }
}

// ---------------------------------------------------------------------------
// Dense WMMA GEMM: D[M,128] = A[M,128] * Wt^T (+bias).  Wt is [128out,128in]
// row-major, i.e. exactly the B-transposed layout the B-fragment wants.
// Block: 256 threads = 8 waves, one n-tile each; the 16x128 f16 A tile is
// DMA'd into LDS by the Tensor Data Mover (wave 0 issues tensor_load_to_lds,
// then s_wait_tensorcnt 0), falling back to a manual copy if the builtin is
// unavailable.  As is the only LDS allocation, so its LDS offset is 0.
// ---------------------------------------------------------------------------
__global__ void k_gemm128(const _Float16* __restrict__ A,
                          const _Float16* __restrict__ Wt,
                          const float* __restrict__ bias,
                          float* __restrict__ oF32, _Float16* __restrict__ oF16,
                          int M) {
  const int tid = threadIdx.x, w = tid >> 5, lane = tid & 31;
  const int half = lane >> 4, ln = lane & 15;
  const int m0 = blockIdx.x * 16;
  __shared__ __align__(16) _Float16 As[16 * 128];
#if HAVE_TDM
  if (tid < 32) {
    // D# group0: count=1 | lds_addr=0 | global_addr | type=2
    unsigned long long ga = (unsigned long long)(const void*)(A + (size_t)m0 * 128);
    u32x4 g0;
    g0[0] = 1u;                                   // count=1, user descriptor
    g0[1] = 0u;                                   // lds_addr (bytes)
    g0[2] = (unsigned)(ga & 0xFFFFFFFFu);         // global_addr[31:0]
    g0[3] = (unsigned)((ga >> 32) & 0x01FFFFFFu)  // global_addr[56:32]
            | (2u << 30);                         // type=2 ("image")
    // D# group1: data_size=2B; tensor 128x16; tile 128x16; stride0=128
    i32x8 g1 = (i32x8)0;
    g1[0] = (1 << 16);           // data_size code 1 => 2 bytes
    g1[1] = (128 << 16);         // tensor_dim0[15:0] @ bits 63:48
    g1[2] = (16 << 16);          // tensor_dim1[15:0] @ bits 95:80
    g1[3] = (128 << 16);         // tile_dim0 @ bits 127:112
    g1[4] = 16;                  // tile_dim1 @ bits 143:128 (tile_dim2=0)
    g1[5] = 128;                 // tensor_dim0_stride[31:0]
    g1[6] = 0;
    g1[7] = 0;
#if defined(__clang_major__) && (__clang_major__ >= 23)
    __builtin_amdgcn_tensor_load_to_lds(g0, g1, (i32x4)0, (i32x4)0, (i32x8)0, 0);
#else
    __builtin_amdgcn_tensor_load_to_lds(g0, g1, (i32x4)0, (i32x4)0, 0);
#endif
    __builtin_amdgcn_s_wait_tensorcnt((short)0);
  }
#else
  for (int e = tid; e < 16 * 128; e += 256) As[e] = A[(size_t)m0 * 128 + e];
#endif
  __syncthreads();
  v8f acc = {};
  const int n0 = w * 16;
#pragma unroll
  for (int k0 = 0; k0 < 128; k0 += 32) {
    v16h af = load_frag(&As[ln * 128 + k0], half);
    v16h bf = load_frag(Wt + (size_t)(n0 + ln) * 128 + k0, half);
    acc = wmma16(af, bf, acc);
  }
#pragma unroll
  for (int r = 0; r < 8; ++r) {
    int m = m0 + r + 8 * half, n = n0 + ln;
    float v = acc[r] + (bias ? bias[n] : 0.f);
    if (oF32) oF32[(size_t)m * 128 + n] = v;
    if (oF16) oF16[(size_t)m * 128 + n] = (_Float16)v;
  }
}

// ---------------------------------------------------------------------------
// NMS-masked scores
// ---------------------------------------------------------------------------
__global__ void k_scores(const float* __restrict__ hm, float* __restrict__ sc) {
  int idx = blockIdx.x * 256 + threadIdx.x;
  if (idx >= NB * NCLS * HW) return;
  int p = idx % HW;
  int c = (idx / HW) % NCLS;
  int y = p / IW, x = p % IW;
  float v = hm[idx];
  float out;
  if (c == 8 || c == 9) out = v;
  else if (y == 0 || y == IH - 1 || x == 0 || x == IW - 1) out = 0.f;
  else {
    const float* hb = hm + (idx - p);
    float mx = v;
#pragma unroll
    for (int dy = -1; dy <= 1; ++dy)
#pragma unroll
      for (int dx = -1; dx <= 1; ++dx)
        mx = fmaxf(mx, hb[(y + dy) * IW + x + dx]);
    out = (v == mx) ? v : 0.f;
  }
  sc[idx] = out;
}

// ---------------------------------------------------------------------------
// Per-batch top-200 (iterative block argmax; lowest-index tie break).
// ---------------------------------------------------------------------------
__global__ void k_topk(float* __restrict__ sc, int* __restrict__ pos,
                       int* __restrict__ cls, float* __restrict__ qpos) {
  int b = blockIdx.x;
  float* s = sc + (size_t)b * NCLS * HW;
  __shared__ unsigned long long red[1024];
  const int t = threadIdx.x;
  for (int it = 0; it < NQ; ++it) {
    unsigned long long best = 0ull;
    for (int i = t; i < NCLS * HW; i += 1024) {
      unsigned long long pk =
          ((unsigned long long)fkey(s[i]) << 32) | (unsigned)(0xFFFFFFFFu - (unsigned)i);
      if (pk > best) best = pk;
    }
    red[t] = best;
    __syncthreads();
    for (int o = 512; o > 0; o >>= 1) {
      if (t < o) red[t] = red[t] > red[t + o] ? red[t] : red[t + o];
      __syncthreads();
    }
    if (t == 0) {
      int idx = (int)(0xFFFFFFFFu - (unsigned)(red[0] & 0xFFFFFFFFull));
      int cc = idx / HW, pp = idx % HW;
      pos[b * NQ + it] = pp;
      cls[b * NQ + it] = cc;
      qpos[(b * NQ + it) * 2 + 0] = (float)(pp / IW) + 0.5f;
      qpos[(b * NQ + it) * 2 + 1] = (float)(pp % IW) + 0.5f;
      s[idx] = -1e30f;
    }
    __syncthreads();
  }
}

// ---------------------------------------------------------------------------
// Query features: qf_t[b,j,h] = ff[b,h,pos] + cls_w[h,cls] + cls_b[h]
// ---------------------------------------------------------------------------
__global__ void k_build_q(const float* __restrict__ ff, const int* __restrict__ pos,
                          const int* __restrict__ cls, const float* __restrict__ cw,
                          const float* __restrict__ cb, float* __restrict__ qf) {
  int idx = blockIdx.x * 256 + threadIdx.x;
  if (idx >= NB * NQP * HIDC) return;
  int h = idx % HIDC, j = (idx / HIDC) % NQP, b = idx / (HIDC * NQP);
  float v = 0.f;
  if (j < NQ) {
    int pp = pos[b * NQ + j], cc = cls[b * NQ + j];
    v = ff[((size_t)b * HIDC + h) * HW + pp] + cw[h * NCLS + cc] + cb[h];
  }
  qf[idx] = v;
}

// ---------------------------------------------------------------------------
// pos_mlp for queries (sp): one block per (j,b), 128 threads
// ---------------------------------------------------------------------------
__global__ void k_posmlp_sp(const float* __restrict__ qpos, const float* __restrict__ w1,
                            const float* __restrict__ b1, const float* __restrict__ s,
                            const float* __restrict__ bb, const float* __restrict__ w2,
                            const float* __restrict__ b2, float* __restrict__ qpe) {
  int j = blockIdx.x, b = blockIdx.y, t = threadIdx.x;
  float py = 0.f, px = 0.f;
  if (j < NQ) { py = qpos[(b * NQ + j) * 2]; px = qpos[(b * NQ + j) * 2 + 1]; }
  __shared__ float h1[HIDC];
  float a = w1[t * 2] * py + w1[t * 2 + 1] * px + b1[t];
  h1[t] = fmaxf(a * s[t] + bb[t], 0.f);
  __syncthreads();
  float o = b2[t];
  for (int i = 0; i < HIDC; ++i) o += h1[i] * w2[t * HIDC + i];
  qpe[((size_t)b * NQP + j) * HIDC + t] = o;
}

// pos_mlp for BEV grid (cp), stage 1 -> f16 h1 matrix [HW,128]
__global__ void k_cp_h1(const float* __restrict__ w1, const float* __restrict__ b1,
                        const float* __restrict__ s, const float* __restrict__ bb,
                        _Float16* __restrict__ h1) {
  int idx = blockIdx.x * 256 + threadIdx.x;
  if (idx >= HW * HIDC) return;
  int t = idx % HIDC, n = idx / HIDC;
  float py = (float)(n / IW) + 0.5f, px = (float)(n % IW) + 0.5f;
  float a = w1[t * 2] * py + w1[t * 2 + 1] * px + b1[t];
  h1[idx] = (_Float16)fmaxf(a * s[t] + bb[t], 0.f);
}

// k_feat f16 = ff^T + kpe
__global__ void k_add_kpe(const float* __restrict__ ff, const float* __restrict__ kpe,
                          _Float16* __restrict__ kf) {
  int idx = blockIdx.x * 256 + threadIdx.x;
  if (idx >= NB * HW * HIDC) return;
  int h = idx % HIDC, n = (idx / HIDC) % HW, b = idx / (HIDC * HW);
  kf[idx] = (_Float16)(ff[((size_t)b * HIDC + h) * HW + n] + kpe[(size_t)n * HIDC + h]);
}

// pack K -> [b][h][key][32] padded, V -> [b][h][d][key] transposed
__global__ void k_pack_kp(const _Float16* __restrict__ Kraw, _Float16* __restrict__ Kp) {
  int idx = blockIdx.x * 256 + threadIdx.x;
  if (idx >= NB * NHEAD * CA_NKP * 32) return;
  int d = idx & 31;
  int key = (idx >> 5) % CA_NKP;
  int h = (idx / (32 * CA_NKP)) % NHEAD;
  int b = idx / (32 * CA_NKP * NHEAD);
  _Float16 v = (_Float16)0.f;
  if (key < HW && d < 16) v = Kraw[((size_t)b * HW + key) * HIDC + h * 16 + d];
  Kp[idx] = v;
}
__global__ void k_pack_vt(const _Float16* __restrict__ Vraw, _Float16* __restrict__ Vt) {
  int idx = blockIdx.x * 256 + threadIdx.x;
  if (idx >= NB * NHEAD * 16 * CA_NKP) return;
  int key = idx % CA_NKP;
  int d = (idx / CA_NKP) % 16;
  int h = (idx / (CA_NKP * 16)) % NHEAD;
  int b = idx / (CA_NKP * 16 * NHEAD);
  _Float16 v = (_Float16)0.f;
  if (key < HW) v = Vraw[((size_t)b * HW + key) * HIDC + h * 16 + d];
  Vt[idx] = v;
}

// self-attn QKV projection from (qf_t + qpe) directly into packed layouts
__global__ void k_sa_qkv(const float* __restrict__ qf, const float* __restrict__ qpe,
                         const float* __restrict__ w, const float* __restrict__ bias,
                         _Float16* __restrict__ Qp, _Float16* __restrict__ Kp,
                         _Float16* __restrict__ Vt) {
  int idx = blockIdx.x * 256 + threadIdx.x;
  if (idx >= NB * 384 * SA_NKP) return;
  int j = idx % SA_NKP, oc = (idx / SA_NKP) % 384, b = idx / (SA_NKP * 384);
  float s = 0.f;
  if (j < NQ) {
    s = bias[oc];
    size_t rb = ((size_t)b * NQP + j) * HIDC;
    for (int i = 0; i < HIDC; ++i) s += (qf[rb + i] + qpe[rb + i]) * w[oc * HIDC + i];
  }
  int which = oc >> 7, hd = oc & 127, h = hd >> 4, d = hd & 15;
  _Float16 v = (_Float16)s;
  if (which == 0) {
    if (j < NQP) {
      size_t o = (((size_t)b * NHEAD + h) * NQP + j) * 32 + d;
      Qp[o] = v; Qp[o + 16] = (_Float16)0.f;
    }
  } else if (which == 1) {
    size_t o = (((size_t)b * NHEAD + h) * SA_NKP + j) * 32 + d;
    Kp[o] = v; Kp[o + 16] = (_Float16)0.f;
  } else {
    Vt[(((size_t)b * NHEAD + h) * 16 + d) * SA_NKP + j] = v;
  }
}

// cross-attn Q projection from (q1 + qpe)
__global__ void k_ca_q(const float* __restrict__ q1, const float* __restrict__ qpe,
                       const float* __restrict__ w, const float* __restrict__ bias,
                       _Float16* __restrict__ Qp) {
  int idx = blockIdx.x * 256 + threadIdx.x;
  if (idx >= NB * 128 * NQP) return;
  int j = idx % NQP, hd = (idx / NQP) % 128, b = idx / (NQP * 128);
  float s = 0.f;
  if (j < NQ) {
    s = bias[hd];
    size_t rb = ((size_t)b * NQP + j) * HIDC;
    for (int i = 0; i < HIDC; ++i) s += (q1[rb + i] + qpe[rb + i]) * w[hd * HIDC + i];
  }
  int h = hd >> 4, d = hd & 15;
  size_t o = (((size_t)b * NHEAD + h) * NQP + j) * 32 + d;
  Qp[o] = (_Float16)s;
  Qp[o + 16] = (_Float16)0.f;
}

// ---------------------------------------------------------------------------
// Flash attention: 1 wave per (b, head, q-tile, key-split).
// Scores: two 16x16x32 WMMAs vs hd-padded K; online softmax with 16-lane
// shuffles; P bounced through LDS (C-layout -> A-layout); PV via WMMA vs V^T.
// ---------------------------------------------------------------------------
__global__ void k_attn_flash(const _Float16* __restrict__ Qp,
                             const _Float16* __restrict__ Kp,
                             const _Float16* __restrict__ Vt, int nqp, int nkp,
                             int validK, int splits, float* __restrict__ opart,
                             float* __restrict__ stats) {
  int bx = blockIdx.x;
  int split = bx % splits;
  int qt = (bx / splits) % QTILES;
  int h = (bx / (splits * QTILES)) % NHEAD;
  int b = bx / (splits * QTILES * NHEAD);
  int lane = threadIdx.x & 31, half = lane >> 4, ln = lane & 15;
  int q0 = qt * 16;
  const _Float16* qrow = Qp + (((size_t)(b * NHEAD + h)) * nqp + q0 + ln) * 32;
  v16h aq = load_frag(qrow, half);
  int nch = nkp / 32;
  int span = (nch + splits - 1) / splits;
  int c0 = split * span;
  int c1 = c0 + span; if (c1 > nch) c1 = nch;
  v8f acc = {};
  float m[8], l[8];
#pragma unroll
  for (int r = 0; r < 8; ++r) { m[r] = -1e30f; l[r] = 0.f; }
  const _Float16* Kb = Kp + (size_t)(b * NHEAD + h) * nkp * 32;
  const _Float16* Vb = Vt + ((size_t)(b * NHEAD + h) * 16) * nkp;
  __shared__ __align__(16) _Float16 Pl[16 * 32];
  for (int ch = c0; ch < c1; ++ch) {
    int kc = ch * 32;
    v8f s0t, s1t;
    {
      v16h bk = load_frag(Kb + (size_t)(kc + ln) * 32, half);
      v8f z = {};
      s0t = wmma16(aq, bk, z);
    }
    {
      v16h bk = load_frag(Kb + (size_t)(kc + 16 + ln) * 32, half);
      v8f z = {};
      s1t = wmma16(aq, bk, z);
    }
    float v0 = (kc + ln < validK) ? 1.f : 0.f;
    float v1 = (kc + 16 + ln < validK) ? 1.f : 0.f;
#pragma unroll
    for (int r = 0; r < 8; ++r) {
      float s0 = v0 > 0.f ? s0t[r] * 0.25f : -1e30f;   // 1/sqrt(16)
      float s1 = v1 > 0.f ? s1t[r] * 0.25f : -1e30f;
      float mt = rmax16(fmaxf(s0, s1));
      float mn = fmaxf(m[r], mt);
      float alpha = __expf(m[r] - mn);
      float p0 = v0 * __expf(s0 - mn);
      float p1 = v1 * __expf(s1 - mn);
      float ps = rsum16(p0 + p1);
      l[r] = l[r] * alpha + ps;
      m[r] = mn;
      acc[r] *= alpha;
      int row = r + 8 * half;
      Pl[row * 32 + ln] = (_Float16)p0;
      Pl[row * 32 + 16 + ln] = (_Float16)p1;
    }
    v16h ap = load_frag(&Pl[ln * 32], half);
    v16h bv = load_frag(Vb + (size_t)ln * nkp + kc, half);
    acc = wmma16(ap, bv, acc);
  }
  size_t pb = (size_t)bx;
#pragma unroll
  for (int r = 0; r < 8; ++r) {
    int row = r + 8 * half;
    opart[pb * 256 + row * 16 + ln] = acc[r];
    stats[pb * 32 + row * 2 + 0] = m[r];
    stats[pb * 32 + row * 2 + 1] = l[r];
  }
}

__global__ void k_attn_combine(const float* __restrict__ opart,
                               const float* __restrict__ stats, float* __restrict__ O,
                               int splits) {
  int bx = blockIdx.x;  // (b*NHEAD+h)*QTILES + qt
  int qt = bx % QTILES, h = (bx / QTILES) % NHEAD, b = bx / (QTILES * NHEAD);
  int t = threadIdx.x, row = t >> 4, col = t & 15;
  size_t base = (size_t)bx * splits;
  float mx = -1e30f;
  for (int s = 0; s < splits; ++s) mx = fmaxf(mx, stats[(base + s) * 32 + row * 2]);
  float den = 0.f, num = 0.f;
  for (int s = 0; s < splits; ++s) {
    float ms = stats[(base + s) * 32 + row * 2];
    float ls = stats[(base + s) * 32 + row * 2 + 1];
    float wgt = __expf(ms - mx);
    den += ls * wgt;
    num += opart[(base + s) * 256 + row * 16 + col] * wgt;
  }
  float v = den > 0.f ? num / den : 0.f;
  O[((size_t)b * NQP + qt * 16 + row) * HIDC + h * 16 + col] = v;
}

// ---------------------------------------------------------------------------
// out-proj + residual + LayerNorm (block of 128 per (b,j))
// ---------------------------------------------------------------------------
__global__ void k_oproj_ln(const float* __restrict__ O, const float* __restrict__ res,
                           const float* __restrict__ ow, const float* __restrict__ ob,
                           const float* __restrict__ g, const float* __restrict__ be,
                           float* __restrict__ out) {
  int j = blockIdx.x % NQP, b = blockIdx.x / NQP, t = threadIdx.x;
  __shared__ float orow[HIDC];
  __shared__ float red[HIDC];
  size_t rb = ((size_t)b * NQP + j) * HIDC;
  orow[t] = O[rb + t];
  __syncthreads();
  float s = ob[t] + res[rb + t];
  for (int i = 0; i < HIDC; ++i) s += orow[i] * ow[t * HIDC + i];
  red[t] = s;
  __syncthreads();
  for (int o = 64; o > 0; o >>= 1) { if (t < o) red[t] += red[t + o]; __syncthreads(); }
  float mu = red[0] / HIDC;
  __syncthreads();
  float d = s - mu;
  red[t] = d * d;
  __syncthreads();
  for (int o = 64; o > 0; o >>= 1) { if (t < o) red[t] += red[t + o]; __syncthreads(); }
  float var = red[0] / HIDC;
  out[rb + t] = d * rsqrtf(var + 1e-5f) * g[t] + be[t];
}

// FFN + residual + LayerNorm (block of 256 per (b,j))
__global__ void k_ffn_ln(const float* __restrict__ q2, const float* __restrict__ w1,
                         const float* __restrict__ b1, const float* __restrict__ w2,
                         const float* __restrict__ b2, const float* __restrict__ g,
                         const float* __restrict__ be, float* __restrict__ out) {
  int j = blockIdx.x % NQP, b = blockIdx.x / NQP, t = threadIdx.x;
  __shared__ float qr[HIDC];
  __shared__ float h1[FFND];
  __shared__ float red[HIDC];
  size_t rb = ((size_t)b * NQP + j) * HIDC;
  if (t < HIDC) qr[t] = q2[rb + t];
  __syncthreads();
  float a = b1[t];
  for (int i = 0; i < HIDC; ++i) a += qr[i] * w1[t * HIDC + i];
  h1[t] = fmaxf(a, 0.f);
  __syncthreads();
  float y = 0.f;
  if (t < HIDC) {
    y = b2[t] + qr[t];
    for (int i = 0; i < FFND; ++i) y += h1[i] * w2[t * FFND + i];
    red[t] = y;
  }
  __syncthreads();
  for (int o = 64; o > 0; o >>= 1) { if (t < o) red[t] += red[t + o]; __syncthreads(); }
  float mu = red[0] / HIDC;
  __syncthreads();
  if (t < HIDC) { float d = y - mu; red[t] = d * d; }
  __syncthreads();
  for (int o = 64; o > 0; o >>= 1) { if (t < o) red[t] += red[t + o]; __syncthreads(); }
  float var = red[0] / HIDC;
  if (t < HIDC) out[rb + t] = (y - mu) * rsqrtf(var + 1e-5f) * g[t] + be[t];
}

// ---------------------------------------------------------------------------
// Separate heads (block of 64 per (b,j))
// ---------------------------------------------------------------------------
struct HeadP { const float *w1, *s, *b, *w2, *b2; int outc, coff; };
struct HeadsArgs { HeadP h[6]; };

__global__ void k_heads(const float* __restrict__ q3, const float* __restrict__ qpos,
                        float* __restrict__ out, HeadsArgs ha) {
  int j = blockIdx.x % NQ, b = blockIdx.x / NQ, t = threadIdx.x;
  __shared__ float qr[HIDC];
  __shared__ float hh[HHID];
  size_t rb = ((size_t)b * NQP + j) * HIDC;
  qr[t] = q3[rb + t];
  qr[t + 64] = q3[rb + t + 64];
  __syncthreads();
  for (int k = 0; k < 6; ++k) {
    HeadP hp = ha.h[k];
    float a = 0.f;
    for (int i = 0; i < HIDC; ++i) a += qr[i] * hp.w1[t * HIDC + i];
    hh[t] = fmaxf(a * hp.s[t] + hp.b[t], 0.f);
    __syncthreads();
    if (t < hp.outc) {
      float v = hp.b2[t];
      for (int i = 0; i < HHID; ++i) v += hh[i] * hp.w2[t * HHID + i];
      if (k == 0) v += qpos[(b * NQ + j) * 2 + t];
      out[((size_t)b * 20 + hp.coff + t) * NQ + j] = v;
    }
    __syncthreads();
  }
}

// ---------------------------------------------------------------------------
// Host launcher
// ---------------------------------------------------------------------------
extern "C" void kernel_launch(void* const* d_in, const int* in_sizes, int n_in,
                              void* d_out, int out_size, void* d_ws, size_t ws_size,
                              hipStream_t stream) {
  const float* IN        = (const float*)d_in[0];
  const float* shared_w  = (const float*)d_in[1];
  const float* shared_b  = (const float*)d_in[2];
  const float* hm_w1     = (const float*)d_in[3];
  const float* hm_bn_s   = (const float*)d_in[4];
  const float* hm_bn_b   = (const float*)d_in[5];
  const float* hm_w2     = (const float*)d_in[6];
  const float* hm_b2     = (const float*)d_in[7];
  const float* cls_w     = (const float*)d_in[8];
  const float* cls_b     = (const float*)d_in[9];
  const float* sp_w1     = (const float*)d_in[10];
  const float* sp_b1     = (const float*)d_in[11];
  const float* sp_bn_s   = (const float*)d_in[12];
  const float* sp_bn_b   = (const float*)d_in[13];
  const float* sp_w2     = (const float*)d_in[14];
  const float* sp_b2     = (const float*)d_in[15];
  const float* cp_w1     = (const float*)d_in[16];
  const float* cp_b1     = (const float*)d_in[17];
  const float* cp_bn_s   = (const float*)d_in[18];
  const float* cp_bn_b   = (const float*)d_in[19];
  const float* cp_w2     = (const float*)d_in[20];
  const float* cp_b2     = (const float*)d_in[21];
  const float* sa_qkv_w  = (const float*)d_in[22];
  const float* sa_qkv_b  = (const float*)d_in[23];
  const float* sa_o_w    = (const float*)d_in[24];
  const float* sa_o_b    = (const float*)d_in[25];
  const float* ca_qkv_w  = (const float*)d_in[26];
  const float* ca_qkv_b  = (const float*)d_in[27];
  const float* ca_o_w    = (const float*)d_in[28];
  const float* ca_o_b    = (const float*)d_in[29];
  const float* ln1_g = (const float*)d_in[30];
  const float* ln1_b = (const float*)d_in[31];
  const float* ln2_g = (const float*)d_in[32];
  const float* ln2_b = (const float*)d_in[33];
  const float* ln3_g = (const float*)d_in[34];
  const float* ln3_b = (const float*)d_in[35];
  const float* ffn_w1 = (const float*)d_in[36];
  const float* ffn_b1 = (const float*)d_in[37];
  const float* ffn_w2 = (const float*)d_in[38];
  const float* ffn_b2 = (const float*)d_in[39];

  size_t cur = 0;
  auto alloc = [&](size_t bytes) -> void* {
    void* p = (char*)d_ws + cur;
    cur += (bytes + 255) & ~(size_t)255;
    return p;
  };

  _Float16* in16    = (_Float16*)alloc((size_t)NB * CINC * HW * 2);
  _Float16* w1_16   = (_Float16*)alloc((size_t)HIDC * CINC * 9 * 2);
  _Float16* w2_16   = (_Float16*)alloc((size_t)HIDC * HIDC * 9 * 2);
  _Float16* w3_16   = (_Float16*)alloc((size_t)NCLS * HIDC * 9 * 2);
  _Float16* cpw2_16 = (_Float16*)alloc((size_t)HIDC * HIDC * 2);
  _Float16* caqkv16 = (_Float16*)alloc((size_t)384 * HIDC * 2);
  float*    ff      = (float*)alloc((size_t)NB * HIDC * HW * 4);
  _Float16* ff16    = (_Float16*)alloc((size_t)NB * HIDC * HW * 2);
  _Float16* h16     = (_Float16*)alloc((size_t)NB * HIDC * HW * 2);
  float*    hm      = (float*)alloc((size_t)NB * NCLS * HW * 4);
  float*    scores  = (float*)alloc((size_t)NB * NCLS * HW * 4);
  int*      pos     = (int*)alloc((size_t)NB * NQ * 4);
  int*      cls     = (int*)alloc((size_t)NB * NQ * 4);
  float*    qpos    = (float*)alloc((size_t)NB * NQ * 2 * 4);
  float*    qf_t    = (float*)alloc((size_t)NB * NQP * HIDC * 4);
  float*    qpe     = (float*)alloc((size_t)NB * NQP * HIDC * 4);
  _Float16* h1cp    = (_Float16*)alloc((size_t)HW * HIDC * 2);
  float*    kpe     = (float*)alloc((size_t)HW * HIDC * 4);
  _Float16* kf16    = (_Float16*)alloc((size_t)NB * HW * HIDC * 2);
  _Float16* Kraw    = (_Float16*)alloc((size_t)NB * HW * HIDC * 2);
  _Float16* Vraw    = (_Float16*)alloc((size_t)NB * HW * HIDC * 2);
  _Float16* Kp_ca   = (_Float16*)alloc((size_t)NB * NHEAD * CA_NKP * 32 * 2);
  _Float16* Vt_ca   = (_Float16*)alloc((size_t)NB * NHEAD * 16 * CA_NKP * 2);
  _Float16* Qp_ca   = (_Float16*)alloc((size_t)NB * NHEAD * NQP * 32 * 2);
  _Float16* Qp_sa   = (_Float16*)alloc((size_t)NB * NHEAD * NQP * 32 * 2);
  _Float16* Kp_sa   = (_Float16*)alloc((size_t)NB * NHEAD * SA_NKP * 32 * 2);
  _Float16* Vt_sa   = (_Float16*)alloc((size_t)NB * NHEAD * 16 * SA_NKP * 2);
  float*    O_sa    = (float*)alloc((size_t)NB * NQP * HIDC * 4);
  float*    O_ca    = (float*)alloc((size_t)NB * NQP * HIDC * 4);
  float*    q1      = (float*)alloc((size_t)NB * NQP * HIDC * 4);
  float*    q2      = (float*)alloc((size_t)NB * NQP * HIDC * 4);
  float*    q3      = (float*)alloc((size_t)NB * NQP * HIDC * 4);
  int maxblk = NB * NHEAD * QTILES * CA_SPLITS;
  float* opart = (float*)alloc((size_t)maxblk * 256 * 4);
  float* stats = (float*)alloc((size_t)maxblk * 32 * 4);

  auto cvt = [&](const float* s, _Float16* d, int n) {
    k_cvt<<<dim3((n + 255) / 256), dim3(256), 0, stream>>>(s, d, n);
  };
  cvt(IN, in16, NB * CINC * HW);
  cvt(shared_w, w1_16, HIDC * CINC * 9);
  cvt(hm_w1, w2_16, HIDC * HIDC * 9);
  cvt(hm_w2, w3_16, NCLS * HIDC * 9);
  cvt(cp_w2, cpw2_16, HIDC * HIDC);
  cvt(ca_qkv_w, caqkv16, 384 * HIDC);

  // conv1: shared 512->128 (+bias) -> ff f32 + ff16
  k_conv_wmma<<<dim3(HW / (16 * CONV_NT), NB), dim3(256), 0, stream>>>(
      in16, w1_16, CINC, 8, 0, shared_b, nullptr, ff, ff16);
  // conv2: 128->128, BN+ReLU -> h16
  k_conv_wmma<<<dim3(HW / (16 * CONV_NT), NB), dim3(256), 0, stream>>>(
      ff16, w2_16, HIDC, 8, 1, hm_bn_s, hm_bn_b, nullptr, h16);
  // conv3: 128->10, +bias, sigmoid -> hm
  k_conv_wmma<<<dim3(HW / (16 * CONV_NT), NB), dim3(32), 0, stream>>>(
      h16, w3_16, HIDC, 1, 2, hm_b2, nullptr, hm, nullptr);

  int nsc = NB * NCLS * HW;
  k_scores<<<dim3((nsc + 255) / 256), dim3(256), 0, stream>>>(hm, scores);
  k_topk<<<dim3(NB), dim3(1024), 0, stream>>>(scores, pos, cls, qpos);

  int nbq = NB * NQP * HIDC;
  k_build_q<<<dim3((nbq + 255) / 256), dim3(256), 0, stream>>>(ff, pos, cls, cls_w,
                                                               cls_b, qf_t);
  k_posmlp_sp<<<dim3(NQP, NB), dim3(128), 0, stream>>>(qpos, sp_w1, sp_b1, sp_bn_s,
                                                       sp_bn_b, sp_w2, sp_b2, qpe);
  int nh1 = HW * HIDC;
  k_cp_h1<<<dim3((nh1 + 255) / 256), dim3(256), 0, stream>>>(cp_w1, cp_b1, cp_bn_s,
                                                             cp_bn_b, h1cp);
  // kpe = h1cp @ cp_w2^T + cp_b2  (WMMA GEMM, M=32400)
  k_gemm128<<<dim3(HW / 16), dim3(256), 0, stream>>>(h1cp, cpw2_16, cp_b2, kpe,
                                                     nullptr, HW);
  int nkf = NB * HW * HIDC;
  k_add_kpe<<<dim3((nkf + 255) / 256), dim3(256), 0, stream>>>(ff, kpe, kf16);
  // cross-attn K/V projections (WMMA GEMM, M = NB*32400)
  k_gemm128<<<dim3(NB * HW / 16), dim3(256), 0, stream>>>(
      kf16, caqkv16 + (size_t)128 * 128, ca_qkv_b + 128, nullptr, Kraw, NB * HW);
  k_gemm128<<<dim3(NB * HW / 16), dim3(256), 0, stream>>>(
      kf16, caqkv16 + (size_t)256 * 128, ca_qkv_b + 256, nullptr, Vraw, NB * HW);
  int npk = NB * NHEAD * CA_NKP * 32;
  k_pack_kp<<<dim3((npk + 255) / 256), dim3(256), 0, stream>>>(Kraw, Kp_ca);
  int npv = NB * NHEAD * 16 * CA_NKP;
  k_pack_vt<<<dim3((npv + 255) / 256), dim3(256), 0, stream>>>(Vraw, Vt_ca);

  // ---- self-attention ----
  int nsq = NB * 384 * SA_NKP;
  k_sa_qkv<<<dim3((nsq + 255) / 256), dim3(256), 0, stream>>>(
      qf_t, qpe, sa_qkv_w, sa_qkv_b, Qp_sa, Kp_sa, Vt_sa);
  k_attn_flash<<<dim3(NB * NHEAD * QTILES * 1), dim3(32), 0, stream>>>(
      Qp_sa, Kp_sa, Vt_sa, NQP, SA_NKP, NQ, 1, opart, stats);
  k_attn_combine<<<dim3(NB * NHEAD * QTILES), dim3(256), 0, stream>>>(opart, stats,
                                                                      O_sa, 1);
  k_oproj_ln<<<dim3(NB * NQP), dim3(128), 0, stream>>>(O_sa, qf_t, sa_o_w, sa_o_b,
                                                       ln1_g, ln1_b, q1);
  // ---- cross-attention ----
  int ncq = NB * 128 * NQP;
  k_ca_q<<<dim3((ncq + 255) / 256), dim3(256), 0, stream>>>(q1, qpe, ca_qkv_w,
                                                            ca_qkv_b, Qp_ca);
  k_attn_flash<<<dim3(NB * NHEAD * QTILES * CA_SPLITS), dim3(32), 0, stream>>>(
      Qp_ca, Kp_ca, Vt_ca, NQP, CA_NKP, HW, CA_SPLITS, opart, stats);
  k_attn_combine<<<dim3(NB * NHEAD * QTILES), dim3(256), 0, stream>>>(opart, stats,
                                                                      O_ca, CA_SPLITS);
  k_oproj_ln<<<dim3(NB * NQP), dim3(128), 0, stream>>>(O_ca, q1, ca_o_w, ca_o_b,
                                                       ln2_g, ln2_b, q2);
  // ---- FFN + LN3 ----
  k_ffn_ln<<<dim3(NB * NQP), dim3(256), 0, stream>>>(q2, ffn_w1, ffn_b1, ffn_w2,
                                                     ffn_b2, ln3_g, ln3_b, q3);
  // ---- heads ----
  HeadsArgs ha;
  const int outc[6] = {2, 1, 3, 2, 2, 10};
  const int coff[6] = {0, 2, 3, 6, 8, 10};
  for (int k = 0; k < 6; ++k) {
    int base = 40 + k * 5;
    ha.h[k].w1 = (const float*)d_in[base + 0];
    ha.h[k].s  = (const float*)d_in[base + 1];
    ha.h[k].b  = (const float*)d_in[base + 2];
    ha.h[k].w2 = (const float*)d_in[base + 3];
    ha.h[k].b2 = (const float*)d_in[base + 4];
    ha.h[k].outc = outc[k];
    ha.h[k].coff = coff[k];
  }
  k_heads<<<dim3(NB * NQ), dim3(64), 0, stream>>>(q3, qpos, (float*)d_out, ha);
}
